// ReservoirComputing_60241211294079
// MI455X (gfx1250) — matrix operand order repaired and checked by reading, not compiled, not run on hardware
//
#include <hip/hip_runtime.h>
#include <hip/hip_bf16.h>

#define T_STEPS 8192
#define INPN    512
#define RESN    2048
#define OUTN    128

#define BETA  0.9f
#define VTH   1.0f
#define ADEC  0.95f
#define AINC  0.1f

typedef __bf16 v16bf __attribute__((ext_vector_type(16)));
typedef float  v8f   __attribute__((ext_vector_type(8)));

// ---------------- workspace layout (bytes) ----------------
#define WS_CNT_OFF   0u          // unsigned counter
#define WS_SPK_OFF   256u        // 2 * RESN floats (double buffered spikes)
#define WS_V_OFF     16640u      // 2 * RESN floats (double buffered membrane)
#define WS_XBF_OFF   33280u      // T*INP bf16  (8 MB)
#define WS_WBF_OFF   (33280u + 8388608u)              // RES*INP bf16 (2 MB)
#define WS_IIN_OFF   (33280u + 8388608u + 2097152u)   // T*RES f32   (64 MB)

// ---------------- helpers ----------------
__device__ __forceinline__ unsigned short f2bf_bits(float f) {
    unsigned u = __builtin_bit_cast(unsigned, f);
    u += 0x7FFFu + ((u >> 16) & 1u);   // round-to-nearest-even
    return (unsigned short)(u >> 16);
}

// zero counter + spike/membrane double buffers (first 33280 bytes of ws)
__global__ void init_state_kernel(float* ws) {
    int i = blockIdx.x * blockDim.x + threadIdx.x;
    if (i < 8320) ws[i] = 0.0f;
}

__global__ void cvt_bf16_kernel(const float* __restrict__ in,
                                unsigned short* __restrict__ out, int n) {
    int i = blockIdx.x * blockDim.x + threadIdx.x;
    if (i < n) out[i] = f2bf_bits(in[i]);
}

// ---------------- Phase 1: Iin = X (T x INP) * W_in^T (INP x RES) ----------------
// One 16x16 output tile per wave; K-loop of 32 with v_wmma_f32_16x16x32_bf16.
// A lane layout (16-bit A 16x32): lane L (L<16) row M=L holds K = 8*half+0..7 (v0..3)
// and K = 16+8*half..+7 (v4..7), half = lane>>4. B (32x16): lane holds col N=lane&15,
// with the same K mapping as A. Both map to two contiguous 16-byte loads.
__global__ __launch_bounds__(256) void icur_gemm_kernel(
        const unsigned short* __restrict__ Xbf,
        const unsigned short* __restrict__ Wbf,
        float* __restrict__ Iin) {
    const int lane = threadIdx.x & 31;
    const int wave = threadIdx.x >> 5;
    const int tile = blockIdx.x * 8 + wave;       // 65536 tiles total
    const int tM   = tile >> 7;                   // 512 row tiles (time)
    const int tN   = tile & 127;                  // 128 col tiles (neurons)
    const int half = lane >> 4;
    const int l    = lane & 15;

    const unsigned short* aBase = Xbf + (size_t)(tM * 16 + l) * INPN;
    const unsigned short* bBase = Wbf + (size_t)(tN * 16 + l) * INPN;

    v8f acc = {};
    for (int k0 = 0; k0 < INPN; k0 += 32) {
        union { uint4 q[2]; v16bf v; } a, b;
        a.q[0] = *(const uint4*)(aBase + k0 + 8 * half);
        a.q[1] = *(const uint4*)(aBase + k0 + 16 + 8 * half);
        b.q[0] = *(const uint4*)(bBase + k0 + 8 * half);
        b.q[1] = *(const uint4*)(bBase + k0 + 16 + 8 * half);
        acc = __builtin_amdgcn_wmma_f32_16x16x32_bf16(
            /*neg_a=*/false, a.v, /*neg_b=*/false, b.v,
            /*c_mod=*/(short)0, acc, /*reuse_a=*/false, /*reuse_b=*/false);
    }
    // C layout: VGPR r -> row r + 8*half, col = lane&15
    float* cBase = Iin + (size_t)(tM * 16 + half * 8) * RESN + tN * 16 + l;
#pragma unroll
    for (int r = 0; r < 8; ++r) cBase[(size_t)r * RESN] = acc[r];
}

// ---------------- Phase 2: persistent sequential scan ----------------
// 128 WGs x 256 threads. WG w owns neurons [16w, 16w+16) and output row w.
// Per step: stage spk_prev -> LDS, 16x2048 matvec from L2-resident W_rec,
// LIF update (threads 0..15 hold v/adapt in registers), publish spk/v to
// parity buffer, monotonic-counter grid barrier, then readout of row w.
#define ROWS_PER_WG 16
#define NWG         (RESN / ROWS_PER_WG)   // 128

__global__ __launch_bounds__(256) void reservoir_scan_kernel(
        const float* __restrict__ Iin,
        const float* __restrict__ Wrec,
        const float* __restrict__ Wro,
        const float* __restrict__ bias,
        float* __restrict__ spkG,          // 2*RESN
        float* __restrict__ vG,            // 2*RESN
        unsigned int* __restrict__ cnt,
        float* __restrict__ out) {
    __shared__ float spkLds[RESN];
    __shared__ float wroLds[RESN];
    __shared__ float recsum[ROWS_PER_WG];
    __shared__ float wredux[8];

    const int tid   = threadIdx.x;
    const int wg    = blockIdx.x;
    const int gbase = wg * ROWS_PER_WG;
    const int r     = tid >> 4;       // row within WG slice (0..15)
    const int j     = tid & 15;       // 16 lanes per row
    const int lane  = tid & 31;
    const int wv    = tid >> 5;

    // cache this WG's readout row once (8 KB LDS)
    for (int c = tid; c < RESN; c += 256) wroLds[c] = Wro[(size_t)wg * RESN + c];

    float v = 0.0f, ad = 0.0f;   // live only in threads 0..15

    for (int t = 0; t < T_STEPS; ++t) {
        const int p  = t & 1;
        const int np = p ^ 1;

        // stage previous spikes
        for (int c = tid; c < RESN; c += 256) spkLds[c] = spkG[p * RESN + c];
        if (t + 1 < T_STEPS)
            __builtin_prefetch(&Iin[(size_t)(t + 1) * RESN + gbase], 0, 0);
        __syncthreads();

        // recurrent matvec: row gbase+r, columns j*128 .. j*128+127
        const float4* wp = (const float4*)(Wrec + (size_t)(gbase + r) * RESN + j * 128);
        const float4* sp = (const float4*)(spkLds + j * 128);
        float s = 0.0f;
#pragma unroll 8
        for (int c4 = 0; c4 < 32; ++c4) {
            float4 w = wp[c4], sv = sp[c4];
            s += w.x * sv.x + w.y * sv.y + w.z * sv.z + w.w * sv.w;
        }
#pragma unroll
        for (int m = 8; m >= 1; m >>= 1) s += __shfl_xor(s, m, 16);
        if (j == 0) recsum[r] = s;
        __syncthreads();

        // LIF update for our 16 neurons
        if (tid < ROWS_PER_WG) {
            const int n = gbase + tid;
            const float cur = Iin[(size_t)t * RESN + n] + recsum[tid];
            v = BETA * v + cur;
            const float thr = VTH + ad;
            const float spk = (v >= thr) ? 1.0f : 0.0f;
            v  -= spk * VTH;                  // soft reset
            ad  = ADEC * ad + AINC * spk;
            spkG[np * RESN + n] = spk;
            vG[np * RESN + n]   = v;
        }
        __syncthreads();

        // grid-wide barrier (monotonic counter, reset each launch by init kernel)
        if (tid == 0) {
            __threadfence();
            atomicAdd(cnt, 1u);
            const unsigned target = (unsigned)gridDim.x * (unsigned)(t + 1);
            while (__hip_atomic_load(cnt, __ATOMIC_RELAXED,
                                     __HIP_MEMORY_SCOPE_AGENT) < target)
                __builtin_amdgcn_s_sleep(2);
            __threadfence();
        }
        __syncthreads();

        // readout: out[t][wg] = dot(Wro[wg,:], v_t) + bias[wg]
        const float* vg = vG + np * RESN;
        float part = 0.0f;
        const int cb = tid * 8;
#pragma unroll
        for (int c = 0; c < 8; ++c) part += wroLds[cb + c] * vg[cb + c];
#pragma unroll
        for (int m = 16; m >= 1; m >>= 1) part += __shfl_xor(part, m, 32);
        if (lane == 0) wredux[wv] = part;
        __syncthreads();
        if (tid == 0) {
            float tot = bias[wg];
#pragma unroll
            for (int i = 0; i < 8; ++i) tot += wredux[i];
            out[(size_t)t * OUTN + wg] = tot;
        }
        __syncthreads();
    }
}

// ---------------- host side ----------------
extern "C" void kernel_launch(void* const* d_in, const int* in_sizes, int n_in,
                              void* d_out, int out_size, void* d_ws, size_t ws_size,
                              hipStream_t stream) {
    const float* x    = (const float*)d_in[0];   // (T,1,INP)
    const float* Win  = (const float*)d_in[1];   // (RES,INP)
    const float* Wrec = (const float*)d_in[2];   // (RES,RES)
    const float* Wro  = (const float*)d_in[3];   // (OUT,RES)
    const float* bias = (const float*)d_in[4];   // (OUT,)
    float* out = (float*)d_out;

    char* ws = (char*)d_ws;
    unsigned int*   cnt  = (unsigned int*)(ws + WS_CNT_OFF);
    float*          spkG = (float*)(ws + WS_SPK_OFF);
    float*          vG   = (float*)(ws + WS_V_OFF);
    unsigned short* xbf  = (unsigned short*)(ws + WS_XBF_OFF);
    unsigned short* wbf  = (unsigned short*)(ws + WS_WBF_OFF);
    float*          Iin  = (float*)(ws + WS_IIN_OFF);

    // 1) zero counter + state double-buffers
    init_state_kernel<<<33, 256, 0, stream>>>((float*)ws);

    // 2) fp32 -> bf16 for GEMM operands
    {
        const int nx = T_STEPS * INPN;
        cvt_bf16_kernel<<<(nx + 255) / 256, 256, 0, stream>>>(x, xbf, nx);
        const int nw = RESN * INPN;
        cvt_bf16_kernel<<<(nw + 255) / 256, 256, 0, stream>>>(Win, wbf, nw);
    }

    // 3) Iin = X * W_in^T via WMMA bf16 (65536 tiles, 8 waves/block)
    icur_gemm_kernel<<<8192, 256, 0, stream>>>(xbf, wbf, Iin);

    // 4) persistent sequential scan
    reservoir_scan_kernel<<<NWG, 256, 0, stream>>>(Iin, Wrec, Wro, bias,
                                                   spkG, vG, cnt, out);
}
